// diff_pool_net1_21371757265532
// MI455X (gfx1250) — matrix operand order repaired and checked by reading, not compiled, not run on hardware
//
#include <hip/hip_runtime.h>

// ---------------------------------------------------------------------------
// DiffPool GNN for MI455X (gfx1250, wave32, WMMA bf16 path)
// ---------------------------------------------------------------------------
typedef __bf16 bf16;
typedef __attribute__((ext_vector_type(16))) __bf16 v16bf;
typedef __attribute__((ext_vector_type(2)))  __bf16 v2bf;
typedef __attribute__((ext_vector_type(8)))  float  v8f;
typedef __attribute__((ext_vector_type(4)))  float  v4f;
typedef __attribute__((ext_vector_type(2)))  float  v2f;
// global-address-space vector pointer: keeps staging loads on the
// global_load_b128 path (LOADcnt only; FLAT would also tick DScnt and
// interfere with the s_wait_dscnt gating of the WMMAs).
typedef __attribute__((address_space(1))) const v4f* gvp;

#define NN   8192
#define EE   262144
#define FF   128
#define HH   128
#define C1   1024
#define C2   128
#define NC   16

constexpr int BM = 128, BN = 128, BK = 32;
constexpr int ASTR = BK + 2, BSTR = BN + 2;

// ---------------- WMMA fragment helpers (wave32 layouts per CDNA5 ISA) ------
union FragU { v16bf v; unsigned u[8]; };

__device__ __forceinline__ v16bf load_a_frag(const bf16* As, int lane, int mbase) {
    const int m = mbase + (lane & 15);
    const int khalf = (lane >> 4) & 1;
    FragU f;
#pragma unroll
    for (int v = 0; v < 8; ++v) {
        const int k0 = ((v >> 2) << 4) + (khalf << 3) + ((v & 3) << 1);
        f.u[v] = *(const unsigned*)(As + m * ASTR + k0);
    }
    return f.v;
}
__device__ __forceinline__ v16bf load_b_frag(const bf16* Bs, int lane, int nbase) {
    const int n = nbase + (lane & 15);
    const int khalf = (lane >> 4) & 1;
    FragU f;
#pragma unroll
    for (int s = 0; s < 16; ++s)
        f.v[s] = Bs[(((s >> 3) << 4) + (khalf << 3) + (s & 7)) * BSTR + n];
    return f.v;
}

// single v_cvt_pk_bf16_f32: pack two floats into one 32-bit bf16 pair
__device__ __forceinline__ unsigned pack_bf16(float a, float b) {
    v2f in; in.x = a; in.y = b;
    union { v2bf v; unsigned u; } t;
    t.v = __builtin_convertvector(in, v2bf);
    return t.u;
}

template <bool TRANS_A>
__device__ __forceinline__ void store_tiles(bf16* As, bf16* Bs, int tid,
                                            const v4f ra[4], const v4f rb[4]) {
#pragma unroll
    for (int i = 0; i < 4; ++i) {
        const int e = tid + i * 256;
        if (TRANS_A) {
            const int m4 = (e & 31) << 2, k = e >> 5;
            As[(m4 + 0) * ASTR + k] = (bf16)ra[i].x;
            As[(m4 + 1) * ASTR + k] = (bf16)ra[i].y;
            As[(m4 + 2) * ASTR + k] = (bf16)ra[i].z;
            As[(m4 + 3) * ASTR + k] = (bf16)ra[i].w;
        } else {
            const int m = e >> 3, k4 = (e & 7) << 2;
            unsigned* p = (unsigned*)(As + m * ASTR + k4);
            p[0] = pack_bf16(ra[i].x, ra[i].y);
            p[1] = pack_bf16(ra[i].z, ra[i].w);
        }
        const int n4 = (e & 31) << 2, k = e >> 5;
        unsigned* q = (unsigned*)(Bs + k * BSTR + n4);
        q[0] = pack_bf16(rb[i].x, rb[i].y);
        q[1] = pack_bf16(rb[i].z, rb[i].w);
    }
}

// ---------------- Tiled WMMA GEMM: C = op(A) @ B (+C) epilogue --------------
// Ping-pong LDS double buffering: one barrier per K-tile; WMMAs of tile t
// overlap global loads + LDS stores of tile t+1. Staging pointers live in
// address_space(1) (global_load_b128, LOADcnt only) and are strength-reduced
// to one 64-bit add per pointer per tile.
// Requires: M % 128 == 0, K % 32 == 0, N % 4 == 0 (all true in this net).
// Epilogue: v = acc (+ C_old if accum); v *= rowscale[m]; v += bias[n]; relu.
template <bool TRANS_A>
__global__ __launch_bounds__(256) void k_gemm_bf16(
    const float* __restrict__ A, int lda, const float* __restrict__ B, int ldb,
    float* __restrict__ C, int ldc, int M, int N, int K,
    const float* __restrict__ bias, const float* __restrict__ rowscale,
    int relu, int accum)
{
    __shared__ alignas(16) bf16 As[2][BM * ASTR];
    __shared__ alignas(16) bf16 Bs[2][BK * BSTR];

    const int tid  = threadIdx.x;
    const int lane = tid & 31;
    const int wave = tid >> 5;      // 8 waves
    const int wm   = wave >> 1;     // 0..3 : 32 rows each
    const int wn   = wave & 1;      // 0..1 : 64 cols each
    const int bm   = blockIdx.y * BM;
    const int bn   = blockIdx.x * BN;

    // per-thread staging pointers in global AS (loop-invariant deltas)
    gvp pa[4];
    gvp pb[4];
    bool bz[4];
#pragma unroll
    for (int i = 0; i < 4; ++i) {
        const int e = tid + i * 256;            // e in [0,1024)
        if (TRANS_A) {                          // A is [K,M]; 4 consecutive m
            const int m4 = (e & 31) << 2, k = e >> 5;
            pa[i] = (gvp)(A + (long)k * lda + bm + m4);
        } else {                                // A is [M,K]; 4 consecutive k
            const int m = e >> 3, k4 = (e & 7) << 2;
            pa[i] = (gvp)(A + (long)(bm + m) * lda + k4);
        }
        const int n4 = (e & 31) << 2, k = e >> 5;
        const int gn = bn + n4;
        const int gnc = gn < N - 3 ? gn : N - 4;   // branchless clamp, in-bounds
        pb[i] = (gvp)(B + (long)k * ldb + gnc);
        bz[i] = (gn >= N);
    }
    const long aStepV = (TRANS_A ? (long)BK * lda : (long)BK) >> 2;  // float4 units
    const long bStepV = ((long)BK * ldb) >> 2;

    v8f acc[2][4] = {};
    v4f ra[4], rb[4];
    const v4f vzero = {0.f, 0.f, 0.f, 0.f};

    // prologue: fetch + stage tile 0
#pragma unroll
    for (int i = 0; i < 4; ++i) {
        ra[i] = *pa[i];  pa[i] += aStepV;
        v4f v = *pb[i];  pb[i] += bStepV;
        if (bz[i]) v = vzero;
        rb[i] = v;
    }
    store_tiles<TRANS_A>(As[0], Bs[0], tid, ra, rb);

    const int T = K / BK;
    for (int t = 0; t < T; ++t) {
        __syncthreads();                         // cur tile visible; nxt free
        const bf16* curA = As[t & 1];
        const bf16* curB = Bs[t & 1];
        bf16* nxtA = As[(t + 1) & 1];
        bf16* nxtB = Bs[(t + 1) & 1];
        const bool more = (t + 1) < T;

        if (more) {                              // global loads overlap compute
#pragma unroll
            for (int i = 0; i < 4; ++i) {
                ra[i] = *pa[i];  pa[i] += aStepV;
                v4f v = *pb[i];  pb[i] += bStepV;
                if (bz[i]) v = vzero;
                rb[i] = v;
            }
        }

        v16bf af[2], bfv[4];
#pragma unroll
        for (int i = 0; i < 2; ++i) af[i] = load_a_frag(curA, lane, wm * 32 + i * 16);
#pragma unroll
        for (int j = 0; j < 4; ++j) bfv[j] = load_b_frag(curB, lane, wn * 64 + j * 16);

        if (more)                                // LDS stores to other buffer
            store_tiles<TRANS_A>(nxtA, nxtB, tid, ra, rb);

#pragma unroll
        for (int i = 0; i < 2; ++i)
#pragma unroll
            for (int j = 0; j < 4; ++j)
                acc[i][j] = __builtin_amdgcn_wmma_f32_16x16x32_bf16(
                    false, af[i], false, bfv[j], (short)0, acc[i][j], false, false);
    }

    // epilogue (C/D layout: vgpr r -> M = r + 8*(lane>=16), N = lane&15)
#pragma unroll
    for (int i = 0; i < 2; ++i)
#pragma unroll
        for (int j = 0; j < 4; ++j)
#pragma unroll
            for (int r = 0; r < 8; ++r) {
                const int m = bm + wm * 32 + i * 16 + r + (((lane >> 4) & 1) << 3);
                const int n = bn + wn * 64 + j * 16 + (lane & 15);
                if (n < N) {
                    float v = acc[i][j][r];
                    if (accum)    v += C[(long)m * ldc + n];
                    if (rowscale) v *= rowscale[m];
                    if (bias)     v += bias[n];
                    if (relu)     v = fmaxf(v, 0.f);
                    C[(long)m * ldc + n] = v;
                }
            }
}

// ---------------- small utility kernels -------------------------------------
__global__ void k_fill(float* p, float v, long n) {
    long i = (long)blockIdx.x * blockDim.x + threadIdx.x;
    if (i < n) p[i] = v;
}
__global__ void k_copy(const float* __restrict__ s, float* __restrict__ d, long n) {
    long i = (long)blockIdx.x * blockDim.x + threadIdx.x;
    if (i < n) d[i] = s[i];
}
__global__ void k_deg_edges(const int* __restrict__ col, const float* __restrict__ ew,
                            float* deg, int E) {
    int e = blockIdx.x * blockDim.x + threadIdx.x;
    if (e < E) atomicAdd(&deg[col[e]], ew[e]);
}
__global__ void k_rsqrt(const float* __restrict__ deg, float* dinv, int n) {
    int i = blockIdx.x * blockDim.x + threadIdx.x;
    if (i < n) { float d = deg[i]; dinv[i] = d > 0.f ? rsqrtf(d) : 0.f; }
}
// self-loop term of sparse GCN: out[i,:] = dinv[i]^2 * h[i,:]  (F = 1<<fsh)
__global__ void k_selfinit(const float* __restrict__ h, const float* __restrict__ dinv,
                           float* __restrict__ out, long total, int fsh) {
    long gid = (long)blockIdx.x * blockDim.x + threadIdx.x;
    if (gid < total) {
        float d = dinv[(int)(gid >> fsh)];
        out[gid] = d * d * h[gid];
    }
}
// edge scatter: out[c,:] += dinv[r]*w*dinv[c]*h[r,:]   (4 floats / thread)
__global__ void k_edge_agg(const int* __restrict__ row, const int* __restrict__ col,
                           const float* __restrict__ ew, const float* __restrict__ dinv,
                           const float* __restrict__ h, float* __restrict__ out,
                           int E, int fsh) {
    const int tsh = fsh - 2;                   // threads-per-edge shift
    long gid = (long)blockIdx.x * blockDim.x + threadIdx.x;
    long total = (long)E << tsh;
    if (gid >= total) return;
    const int e = (int)(gid >> tsh);
    const int f = ((int)gid & ((1 << tsh) - 1)) << 2;
    const int r = row[e], c = col[e];
    const float nrm = dinv[r] * ew[e] * dinv[c];
    const float4 hv = *(const float4*)(h + ((long)r << fsh) + f);
    float* o = out + ((long)c << fsh) + f;
    atomicAdd(o + 0, nrm * hv.x);
    atomicAdd(o + 1, nrm * hv.y);
    atomicAdd(o + 2, nrm * hv.z);
    atomicAdd(o + 3, nrm * hv.w);
}
__global__ void k_bias_relu(float* x, const float* __restrict__ b, long n, int fmask, int relu) {
    long gid = (long)blockIdx.x * blockDim.x + threadIdx.x;
    if (gid < n) {
        float v = x[gid] + b[(int)gid & fmask];
        x[gid] = relu ? fmaxf(v, 0.f) : v;
    }
}
__global__ void k_scale_rows(const float* __restrict__ in, const float* __restrict__ s,
                             float* __restrict__ out, long n, int fsh) {
    long gid = (long)blockIdx.x * blockDim.x + threadIdx.x;
    if (gid < n) out[gid] = s[(int)(gid >> fsh)] * in[gid];
}
// block-per-row softmax + entropy accumulation (sum over rows of -p*log(p+eps))
__global__ __launch_bounds__(256) void k_softmax_ent(const float* __restrict__ X,
                                                     float* __restrict__ P,
                                                     float* ent_acc, int F) {
    __shared__ float red[256];
    const int rowi = blockIdx.x, tid = threadIdx.x;
    const float* x = X + (long)rowi * F;
    float* p = P + (long)rowi * F;
    float mx = -3.4e38f;
    for (int f = tid; f < F; f += 256) mx = fmaxf(mx, x[f]);
    red[tid] = mx; __syncthreads();
    for (int s = 128; s > 0; s >>= 1) { if (tid < s) red[tid] = fmaxf(red[tid], red[tid + s]); __syncthreads(); }
    mx = red[0]; __syncthreads();
    float sum = 0.f;
    for (int f = tid; f < F; f += 256) sum += expf(x[f] - mx);
    red[tid] = sum; __syncthreads();
    for (int s = 128; s > 0; s >>= 1) { if (tid < s) red[tid] += red[tid + s]; __syncthreads(); }
    const float inv = 1.f / red[0]; __syncthreads();
    float ent = 0.f;
    for (int f = tid; f < F; f += 256) {
        float pv = expf(x[f] - mx) * inv;
        p[f] = pv;
        ent -= pv * logf(pv + 1e-15f);
    }
    red[tid] = ent; __syncthreads();
    for (int s = 128; s > 0; s >>= 1) { if (tid < s) red[tid] += red[tid + s]; __syncthreads(); }
    if (tid == 0) atomicAdd(ent_acc, red[0]);
}
__global__ void k_logsoftmax(const float* __restrict__ X, float* __restrict__ O, int M, int F) {
    int i = blockIdx.x * blockDim.x + threadIdx.x;
    if (i >= M) return;
    const float* x = X + (long)i * F;
    float* o = O + (long)i * F;
    float mx = -3.4e38f;
    for (int f = 0; f < F; ++f) mx = fmaxf(mx, x[f]);
    float s = 0.f;
    for (int f = 0; f < F; ++f) s += expf(x[f] - mx);
    const float l = logf(s);
    for (int f = 0; f < F; ++f) o[f] = x[f] - mx - l;
}
__global__ void k_diagfix(const float* __restrict__ A, float* __restrict__ Ah, int n) {
    long gid = (long)blockIdx.x * blockDim.x + threadIdx.x;
    if (gid < (long)n * n) {
        int i = (int)(gid / n), j = (int)(gid % n);
        float v = A[gid];
        if (i == j && v == 0.f) v = 1.f;
        Ah[gid] = v;
    }
}
__global__ void k_colsum(const float* __restrict__ A, float* deg, int n) {
    int j = blockIdx.x * blockDim.x + threadIdx.x;
    if (j < n) {
        float s = 0.f;
        for (int i = 0; i < n; ++i) s += A[(long)i * n + j];
        deg[j] = s;
    }
}
__global__ __launch_bounds__(256) void k_sumsq(const float* __restrict__ x, long n, float* acc) {
    __shared__ float red[256];
    long i = (long)blockIdx.x * blockDim.x + threadIdx.x;
    const long stride = (long)gridDim.x * blockDim.x;
    float s = 0.f;
    for (; i < n; i += stride) { float v = x[i]; s += v * v; }
    red[threadIdx.x] = s; __syncthreads();
    for (int t = 128; t > 0; t >>= 1) { if (threadIdx.x < t) red[threadIdx.x] += red[threadIdx.x + t]; __syncthreads(); }
    if (threadIdx.x == 0) atomicAdd(acc, red[0]);
}
__global__ void k_trace(const float* __restrict__ A, int n, float* acc) {
    int i = blockIdx.x * blockDim.x + threadIdx.x;
    if (i < n) atomicAdd(acc, A[(long)i * n + i]);
}
// edge_loss = l1 + e1 + l2 + e2 using Frobenius expansion (SS^T never formed)
__global__ void k_loss(const float* __restrict__ sc, float* out) {
    float l1 = sqrtf(fmaxf(sc[0] - 2.f * sc[1] + sc[2], 0.f)) / (8192.f * 8192.f);
    float e1 = sc[3] / 8192.f;
    float l2 = sqrtf(fmaxf(sc[4] - 2.f * sc[5] + sc[6], 0.f)) / (1024.f * 1024.f);
    float e2 = sc[7] / 1024.f;
    out[0] = l1 + e1 + l2 + e2;
}

// ---------------- host-side helpers -----------------------------------------
static inline long cdivl(long a, long b) { return (a + b - 1) / b; }

static void gemm(hipStream_t s, bool transA, const float* A, int lda, const float* B, int ldb,
                 float* C, int ldc, int M, int N, int K,
                 const float* bias, const float* rowscale, int relu, int accum) {
    dim3 g((unsigned)cdivl(N, 128), (unsigned)cdivl(M, 128)), b(256);
    if (transA)
        k_gemm_bf16<true><<<g, b, 0, s>>>(A, lda, B, ldb, C, ldc, M, N, K, bias, rowscale, relu, accum);
    else
        k_gemm_bf16<false><<<g, b, 0, s>>>(A, lda, B, ldb, C, ldc, M, N, K, bias, rowscale, relu, accum);
}

// workspace layout (float offsets; all 16-byte aligned)
#define WS_X0H   0L                       // x0_            8192*128
#define WS_BIG   1048576L                 // hp1 then t1    8192*1024
#define WS_SOFT1 9437184L                 // softmax(s1)    8192*1024
#define WS_G1    17825792L                // s1^T s1        1024*1024
#define WS_H0    18874368L                // h0 then up1    8192*128
#define WS_ADJ1H 19922944L                // adj1 + diagfix 1024*1024
#define WS_ADJ1  20971520L                // adj1 (aligned) 1024*1024
#define WS_H0O   22020096L                // 8192*16
#define WS_X0OUT 22151168L                // 8192*16
#define WS_X1    22282240L                // 1024*128
#define WS_H1    22413312L
#define WS_HD    22544384L
#define WS_X1_   22675456L
#define WS_S2    22806528L
#define WS_SOFT2 22937600L
#define WS_T2    23068672L
#define WS_UP2   23199744L
#define WS_X1OUT 23330816L
#define WS_X2    23461888L                // 128*128
#define WS_H2    23478272L
#define WS_HD2   23494656L
#define WS_ADJ2  23511040L
#define WS_ADJ2H 23527424L
#define WS_X2OUT 23543808L
#define WS_G2    23560192L
#define WS_DEG0  23576576L                // 8192
#define WS_DINV0 23584768L                // 8192
#define WS_DEG1  23592960L                // 1024
#define WS_DINV1 23593984L
#define WS_DEG2  23595008L                // 128
#define WS_DINV2 23595136L
#define WS_SC    23595264L                // 16 scalar accumulators

extern "C" void kernel_launch(void* const* d_in, const int* in_sizes, int n_in,
                              void* d_out, int out_size, void* d_ws, size_t ws_size,
                              hipStream_t stream) {
    const float* x0     = (const float*)d_in[0];
    const int*   eidx   = (const int*)d_in[1];
    const int*   row    = eidx;
    const int*   col    = eidx + EE;
    const float* ew     = (const float*)d_in[2];
    const float* adj    = (const float*)d_in[3];
    const float* W0_in  = (const float*)d_in[5];
    const float* b0_in  = (const float*)d_in[6];
    const float* Wp1    = (const float*)d_in[7];
    const float* bp1    = (const float*)d_in[8];
    const float* W1_in  = (const float*)d_in[9];
    const float* b1_in  = (const float*)d_in[10];
    const float* Wp2    = (const float*)d_in[11];
    const float* bp2    = (const float*)d_in[12];
    const float* W2_in  = (const float*)d_in[13];
    const float* b2_in  = (const float*)d_in[14];
    const float* W1_out = (const float*)d_in[15];
    const float* b1_out = (const float*)d_in[16];
    const float* W0_out = (const float*)d_in[17];
    const float* b0_out = (const float*)d_in[18];

    float* ws = (float*)d_ws;
    float* out_pred = (float*)d_out;                         // 8192*16
    float* out_s1   = out_pred + (long)NN * NC;              // 8192*1024
    float* out_loss = out_s1 + (long)NN * C1;                // scalar
    float* out_adj1 = out_loss + 1;                          // 1024*1024 (odd offset!)

    float* x0_ = ws + WS_X0H;  float* big = ws + WS_BIG;   float* soft1 = ws + WS_SOFT1;
    float* G1  = ws + WS_G1;   float* h0  = ws + WS_H0;    float* adj1h = ws + WS_ADJ1H;
    float* adj1 = ws + WS_ADJ1;
    float* h0o = ws + WS_H0O;  float* x0o = ws + WS_X0OUT; float* x1    = ws + WS_X1;
    float* h1  = ws + WS_H1;   float* hd  = ws + WS_HD;    float* x1_   = ws + WS_X1_;
    float* s2  = ws + WS_S2;   float* soft2 = ws + WS_SOFT2; float* t2  = ws + WS_T2;
    float* up2 = ws + WS_UP2;  float* x1out = ws + WS_X1OUT;
    float* x2  = ws + WS_X2;   float* h2  = ws + WS_H2;    float* hd2 = ws + WS_HD2;
    float* adj2 = ws + WS_ADJ2; float* adj2h = ws + WS_ADJ2H; float* x2out = ws + WS_X2OUT;
    float* G2 = ws + WS_G2;
    float* deg0 = ws + WS_DEG0; float* dinv0 = ws + WS_DINV0;
    float* deg1 = ws + WS_DEG1; float* dinv1 = ws + WS_DINV1;
    float* deg2 = ws + WS_DEG2; float* dinv2 = ws + WS_DINV2;
    float* sc = ws + WS_SC;

    const dim3 b256(256);
#define G1D(n) dim3((unsigned)cdivl((long)(n), 256))

    // ---- sparse GCN degrees: deg = 1 (self loop) + segsum(ew @ col)
    k_fill<<<G1D(NN), b256, 0, stream>>>(deg0, 1.f, NN);
    k_fill<<<1, 16, 0, stream>>>(sc, 0.f, 16);
    k_deg_edges<<<G1D(EE), b256, 0, stream>>>(col, ew, deg0, EE);
    k_rsqrt<<<G1D(NN), b256, 0, stream>>>(deg0, dinv0, NN);

    // ---- x0_ = relu(gcn_sparse(x0, W0_in) + b0_in)
    gemm(stream, false, x0, FF, W0_in, HH, h0, HH, NN, HH, FF, nullptr, nullptr, 0, 0);
    k_selfinit<<<G1D((long)NN * HH), b256, 0, stream>>>(h0, dinv0, x0_, (long)NN * HH, 7);
    k_edge_agg<<<G1D((long)EE * (HH / 4)), b256, 0, stream>>>(row, col, ew, dinv0, h0, x0_, EE, 7);
    k_bias_relu<<<G1D((long)NN * HH), b256, 0, stream>>>(x0_, b0_in, (long)NN * HH, HH - 1, 1);

    // ---- s1 = relu(gcn_sparse(x0_, Wp1) + bp1)   (written to d_out)
    gemm(stream, false, x0_, HH, Wp1, C1, big, C1, NN, C1, HH, nullptr, nullptr, 0, 0);
    k_selfinit<<<G1D((long)NN * C1), b256, 0, stream>>>(big, dinv0, out_s1, (long)NN * C1, 10);
    k_edge_agg<<<G1D((long)EE * (C1 / 4)), b256, 0, stream>>>(row, col, ew, dinv0, big, out_s1, EE, 10);
    k_bias_relu<<<G1D((long)NN * C1), b256, 0, stream>>>(out_s1, bp1, (long)NN * C1, C1 - 1, 1);

    // ---- diff_pool level 1
    k_softmax_ent<<<dim3(NN), b256, 0, stream>>>(out_s1, soft1, &sc[3], C1);
    gemm(stream, false, adj, NN, soft1, C1, big, C1, NN, C1, NN, nullptr, nullptr, 0, 0);      // t1 = adj@S1
    gemm(stream, true, soft1, C1, big, C1, adj1, C1, C1, C1, NN, nullptr, nullptr, 0, 0);      // adj1 = S1^T t1
    k_copy<<<G1D((long)C1 * C1), b256, 0, stream>>>(adj1, out_adj1, (long)C1 * C1);
    gemm(stream, true, soft1, C1, x0_, HH, x1, HH, C1, HH, NN, nullptr, nullptr, 0, 0);        // x1 = S1^T x0_
    gemm(stream, true, soft1, C1, soft1, C1, G1, C1, C1, C1, NN, nullptr, nullptr, 0, 0);      // G1 = S1^T S1
    k_sumsq<<<dim3(4096), b256, 0, stream>>>(adj, (long)NN * NN, &sc[0]);
    k_trace<<<G1D(C1), b256, 0, stream>>>(adj1, C1, &sc[1]);
    k_sumsq<<<dim3(1024), b256, 0, stream>>>(G1, (long)C1 * C1, &sc[2]);

    // ---- dense GCN normalization for adj1
    k_diagfix<<<G1D((long)C1 * C1), b256, 0, stream>>>(adj1, adj1h, C1);
    k_colsum<<<G1D(C1), b256, 0, stream>>>(adj1h, deg1, C1);
    k_rsqrt<<<G1D(C1), b256, 0, stream>>>(deg1, dinv1, C1);

    // ---- x1_ = relu(D1 adj1h^T D1 (x1@W1_in) + b1_in)
    gemm(stream, false, x1, HH, W1_in, HH, h1, HH, C1, HH, HH, nullptr, nullptr, 0, 0);
    k_scale_rows<<<G1D((long)C1 * HH), b256, 0, stream>>>(h1, dinv1, hd, (long)C1 * HH, 7);
    gemm(stream, true, adj1h, C1, hd, HH, x1_, HH, C1, HH, C1, b1_in, dinv1, 1, 0);
    // ---- s2 = relu(D1 adj1h^T D1 (x1_@Wp2) + bp2)
    gemm(stream, false, x1_, HH, Wp2, C2, h1, C2, C1, C2, HH, nullptr, nullptr, 0, 0);
    k_scale_rows<<<G1D((long)C1 * C2), b256, 0, stream>>>(h1, dinv1, hd, (long)C1 * C2, 7);
    gemm(stream, true, adj1h, C1, hd, C2, s2, C2, C1, C2, C1, bp2, dinv1, 1, 0);

    // ---- diff_pool level 2
    k_softmax_ent<<<dim3(C1), b256, 0, stream>>>(s2, soft2, &sc[7], C2);
    gemm(stream, true, soft2, C2, x1_, HH, x2, HH, C2, HH, C1, nullptr, nullptr, 0, 0);        // x2
    gemm(stream, false, adj1, C1, soft2, C2, t2, C2, C1, C2, C1, nullptr, nullptr, 0, 0);      // t2 = adj1@S2
    gemm(stream, true, soft2, C2, t2, C2, adj2, C2, C2, C2, C1, nullptr, nullptr, 0, 0);       // adj2
    gemm(stream, true, soft2, C2, soft2, C2, G2, C2, C2, C2, C1, nullptr, nullptr, 0, 0);      // G2
    k_sumsq<<<dim3(1024), b256, 0, stream>>>(adj1, (long)C1 * C1, &sc[4]);
    k_trace<<<G1D(C2), b256, 0, stream>>>(adj2, C2, &sc[5]);
    k_sumsq<<<dim3(64), b256, 0, stream>>>(G2, (long)C2 * C2, &sc[6]);

    // ---- level-2 dense GCN
    k_diagfix<<<G1D((long)C2 * C2), b256, 0, stream>>>(adj2, adj2h, C2);
    k_colsum<<<G1D(C2), b256, 0, stream>>>(adj2h, deg2, C2);
    k_rsqrt<<<G1D(C2), b256, 0, stream>>>(deg2, dinv2, C2);
    gemm(stream, false, x2, HH, W2_in, HH, h2, HH, C2, HH, HH, nullptr, nullptr, 0, 0);
    k_scale_rows<<<G1D((long)C2 * HH), b256, 0, stream>>>(h2, dinv2, hd2, (long)C2 * HH, 7);
    gemm(stream, true, adj2h, C2, hd2, HH, x2out, HH, C2, HH, C2, b2_in, dinv2, 1, 0);

    // ---- unpool 2 + level-1 output GCN (concat via accumulating GEMMs)
    gemm(stream, false, s2, C2, x2out, HH, up2, HH, C1, HH, C2, nullptr, nullptr, 0, 0);       // up2 = s2@x2_out
    gemm(stream, false, x1_, HH, W1_out, HH, h1, HH, C1, HH, HH, nullptr, nullptr, 0, 0);
    gemm(stream, false, up2, HH, W1_out + (long)HH * HH, HH, h1, HH, C1, HH, HH, nullptr, nullptr, 0, 1);
    k_scale_rows<<<G1D((long)C1 * HH), b256, 0, stream>>>(h1, dinv1, hd, (long)C1 * HH, 7);
    gemm(stream, true, adj1h, C1, hd, HH, x1out, HH, C1, HH, C1, b1_out, dinv1, 1, 0);

    // ---- unpool 1 + output sparse GCN
    gemm(stream, false, out_s1, C1, x1out, HH, h0, HH, NN, HH, C1, nullptr, nullptr, 0, 0);    // up1 (reuse h0)
    gemm(stream, false, x0_, HH, W0_out, NC, h0o, NC, NN, NC, HH, nullptr, nullptr, 0, 0);
    gemm(stream, false, h0, HH, W0_out + (long)HH * NC, NC, h0o, NC, NN, NC, HH, nullptr, nullptr, 0, 1);
    k_selfinit<<<G1D((long)NN * NC), b256, 0, stream>>>(h0o, dinv0, x0o, (long)NN * NC, 4);
    k_edge_agg<<<G1D((long)EE * (NC / 4)), b256, 0, stream>>>(row, col, ew, dinv0, h0o, x0o, EE, 4);
    k_bias_relu<<<G1D((long)NN * NC), b256, 0, stream>>>(x0o, b0_out, (long)NN * NC, NC - 1, 1);
    k_logsoftmax<<<G1D(NN), b256, 0, stream>>>(x0o, out_pred, NN, NC);

    // ---- combined edge loss
    k_loss<<<1, 1, 0, stream>>>(sc, out_loss);
}
